// TransformerEncoder_71141838291875
// MI455X (gfx1250) — compile-verified, matrix operand order checked by reading, not run on hardware
//
#include <hip/hip_runtime.h>
#include <hip/hip_bf16.h>
#include <math.h>

typedef unsigned short u16;
typedef unsigned int   u32;
typedef __attribute__((ext_vector_type(16))) __bf16 v16bf;
typedef __attribute__((ext_vector_type(8)))  float  v8f;

#define N_B    128
#define N_Q    32
#define N_F    7
#define N_MD   128
#define N_H    4
#define N_L    6
#define N_FFN  2048
#define N_TD   32
#define N_S    8
#define N_D    512
#define N_BQ   (N_B * N_Q)      /* 4096  */
#define N_ROWS (N_BQ * N_S)     /* 32768 */

// ---------- bf16 helpers (bit-level, round-to-nearest-even) ----------
__device__ __forceinline__ float bf2f(u16 h) {
    return __uint_as_float(((u32)h) << 16);
}
__device__ __forceinline__ u16 f2bf(float f) {
    u32 u = __float_as_uint(f);
    u32 r = (u + 0x7FFFu + ((u >> 16) & 1u)) >> 16;
    return (u16)r;
}
__device__ __forceinline__ float gelu_tanh(float x) {
    float x3 = x * x * x;
    return 0.5f * x * (1.0f + tanhf(0.7978845608028654f * (x + 0.044715f * x3)));
}

// ---------------------------------------------------------------------
// Tiled bf16 WMMA GEMM:  C[M,N] = act(A[M,K] @ Wt[N,K]^T + bias (+ res))
// Wt is the PRE-TRANSPOSED weight, row-major [N][K] (ldw = K), so both
// A and B tiles stage into LDS with contiguous 128b loads / wide DS
// stores.  block = 256 threads = 8 waves; block tile 128x64; wave tile
// 32x32 (2x2 tiles of v_wmma_f32_16x16x32_bf16); K staged 32-wide.
// ---------------------------------------------------------------------
template<int ACT, bool RES, bool OUTF32>
__launch_bounds__(256)
__global__ void gemm_bf16_wmma(const u16* __restrict__ A, int lda,
                               const u16* __restrict__ Wt, int ldw,
                               const float* __restrict__ bias,
                               const u16* __restrict__ res, int ldr,
                               void* __restrict__ Cout, int ldc,
                               int M, int N, int K)
{
    __shared__ __align__(16) u16 sA[128 * 36];   // [m][k], k-pad to 36
    __shared__ __align__(16) u16 sB[64 * 36];    // [n][k]

    const int tid   = threadIdx.x;
    const int lane  = tid & 31;
    const int wv    = tid >> 5;
    const int waveM = wv >> 1;      // 0..3
    const int waveN = wv & 1;       // 0..1
    const int lm    = lane & 15;
    const int lh    = lane >> 4;    // K-half selector per ISA A-layout
    const int Mg    = blockIdx.y * 128;
    const int Ng    = blockIdx.x * 64;

    union Acc { v8f v; float f[8]; };
    Acc acc[2][2];
#pragma unroll
    for (int mi = 0; mi < 2; ++mi)
#pragma unroll
        for (int ni = 0; ni < 2; ++ni)
#pragma unroll
            for (int r = 0; r < 8; ++r) acc[mi][ni].f[r] = 0.0f;

    const int arow = tid >> 1;            // 0..127
    const int akc  = (tid & 1) * 16;      // 0 or 16
    const int brow = tid >> 2;            // 0..63
    const int bkc  = (tid & 3) * 8;       // 0,8,16,24

    for (int kk = 0; kk < K; kk += 32) {
        // ---- stage A tile 128x32 (each thread: 16 contiguous bf16) ----
        {
            const uint4* ga = (const uint4*)(A + (size_t)(Mg + arow) * lda + kk + akc);
            uint4 a0 = ga[0], a1 = ga[1];
            u32* da = (u32*)&sA[arow * 36 + akc];
            da[0] = a0.x; da[1] = a0.y; da[2] = a0.z; da[3] = a0.w;
            da[4] = a1.x; da[5] = a1.y; da[6] = a1.z; da[7] = a1.w;
        }
        // ---- stage B tile 64x32 from pre-transposed Wt[N][K] ----
        {
            const uint4* gb = (const uint4*)(Wt + (size_t)(Ng + brow) * ldw + kk + bkc);
            uint4 b0 = gb[0];
            u32* db = (u32*)&sB[brow * 36 + bkc];
            db[0] = b0.x; db[1] = b0.y; db[2] = b0.z; db[3] = b0.w;
        }
        __syncthreads();

        // ---- build fragments per ISA 16-bit A layout, run 4 WMMAs ----
        union Frag { v16bf v; u32 u[8]; };
        Frag fa[2], fb[2];
#pragma unroll
        for (int mi = 0; mi < 2; ++mi) {
            const u16* rowp = &sA[(waveM * 32 + mi * 16 + lm) * 36];
#pragma unroll
            for (int r = 0; r < 8; ++r) {
                int k = (r < 4) ? (lh * 8 + 2 * r) : (16 + lh * 8 + 2 * (r - 4));
                fa[mi].u[r] = *(const u32*)&rowp[k];
            }
        }
#pragma unroll
        for (int ni = 0; ni < 2; ++ni) {
            const u16* rowp = &sB[(waveN * 32 + ni * 16 + lm) * 36];
#pragma unroll
            for (int r = 0; r < 8; ++r) {
                int k = (r < 4) ? (lh * 8 + 2 * r) : (16 + lh * 8 + 2 * (r - 4));
                fb[ni].u[r] = *(const u32*)&rowp[k];
            }
        }
#pragma unroll
        for (int mi = 0; mi < 2; ++mi)
#pragma unroll
            for (int ni = 0; ni < 2; ++ni)
                acc[mi][ni].v = __builtin_amdgcn_wmma_f32_16x16x32_bf16(
                    false, fa[mi].v, false, fb[ni].v,
                    (short)0, acc[mi][ni].v, false, false);
        __syncthreads();
    }

    // ---- epilogue: D layout (lane<16: N=lane, M=r; lane>=16: M=8+r) ----
#pragma unroll
    for (int mi = 0; mi < 2; ++mi)
#pragma unroll
        for (int ni = 0; ni < 2; ++ni) {
            int ncol  = Ng + waveN * 32 + ni * 16 + lm;
            int mbase = Mg + waveM * 32 + mi * 16 + lh * 8;
            float bv = bias[ncol];
#pragma unroll
            for (int r = 0; r < 8; ++r) {
                int m = mbase + r;
                float v = acc[mi][ni].f[r] + bv;
                if (RES) v += bf2f(res[(size_t)m * ldr + ncol]);
                if (ACT == 1) v = gelu_tanh(v);
                if (OUTF32) ((float*)Cout)[(size_t)m * ldc + ncol] = v;
                else        ((u16*)Cout)[(size_t)m * ldc + ncol] = f2bf(v);
            }
        }
}

// ---------------------------------------------------------------------
// Misc small kernels
// ---------------------------------------------------------------------
// (L,K,N) f32 -> (L,N,K) bf16 (one-time per launch; cost negligible)
__global__ void k_w_transpose_bf16(const float* __restrict__ src, u16* __restrict__ dst,
                                   int K, int N, int total)
{
    int i = blockIdx.x * blockDim.x + threadIdx.x;
    if (i >= total) return;
    int per = K * N;
    int l = i / per, rem = i - l * per;
    int k = rem / N, n = rem - k * N;
    dst[(size_t)l * per + (size_t)n * K + k] = f2bf(src[i]);
}

// x0[bq, s, d] = (s==0 ? cls : support_emb[b, s-1])[d % MD]
__global__ void k_build_x0(const float* __restrict__ sup, const float* __restrict__ cls,
                           u16* __restrict__ X)
{
    size_t idx = (size_t)blockIdx.x * blockDim.x + threadIdx.x;
    if (idx >= (size_t)N_ROWS * N_D) return;
    int d = (int)(idx & (N_D - 1));
    size_t row = idx >> 9;           // bq*8 + s
    int s = (int)(row & 7);
    size_t bq = row >> 3;
    int b = (int)(bq >> 5);          // Q = 32
    int md = d & (N_MD - 1);
    float v = (s == 0) ? cls[md] : sup[((size_t)b * N_F + (s - 1)) * N_MD + md];
    X[idx] = f2bf(v);
}

// bias[l][bq][s][t] = sum_td cos((t_s - t_t)*freq + phase) * time_w[l]
__launch_bounds__(256)
__global__ void k_bias(const float* __restrict__ query_t, const float* __restrict__ support_t,
                       const float* __restrict__ freq, const float* __restrict__ phase,
                       const float* __restrict__ time_w, float* __restrict__ Bias)
{
    int idx = blockIdx.x * blockDim.x + threadIdx.x;
    if (idx >= N_BQ * 64) return;
    int bq = idx >> 6;
    int s = (idx >> 3) & 7, t = idx & 7;
    int b = bq >> 5, qi = bq & 31;
    float qt = query_t[b * N_Q + qi];
    float ts = (s == 0) ? qt : support_t[b * N_F + s - 1];
    float tt = (t == 0) ? qt : support_t[b * N_F + t - 1];
    float dt = ts - tt;
    float c[N_TD];
#pragma unroll
    for (int j = 0; j < N_TD; ++j) c[j] = cosf(dt * freq[j] + phase[j]);
    for (int l = 0; l < N_L; ++l) {
        float sum = 0.f;
#pragma unroll
        for (int j = 0; j < N_TD; ++j) sum += c[j] * time_w[l * N_TD + j];
        Bias[(size_t)l * N_BQ * 64 + idx] = sum;
    }
}

// One block (64 thr) per contiguous 1024-elem "head" chunk (raw .view()
// semantics => each chunk is row-major [8][128]); bias row = b2 % BQ.
__launch_bounds__(64)
__global__ void k_attention(const u16* __restrict__ Qb, const u16* __restrict__ Kb,
                            const u16* __restrict__ Vb, const float* __restrict__ biasL,
                            u16* __restrict__ Ctx)
{
    __shared__ __align__(16) u16 sq[1024], sk[1024], sv[1024];
    __shared__ float sc[64];
    const int tid = threadIdx.x;
    const size_t base = (size_t)blockIdx.x * 1024;
    {
        const uint4* gq = (const uint4*)(Qb + base) + tid * 2;
        const uint4* gk = (const uint4*)(Kb + base) + tid * 2;
        const uint4* gv = (const uint4*)(Vb + base) + tid * 2;
        uint4 q0 = gq[0], q1 = gq[1], k0 = gk[0], k1 = gk[1], v0 = gv[0], v1 = gv[1];
        u32* dq = (u32*)&sq[tid * 16];
        dq[0]=q0.x; dq[1]=q0.y; dq[2]=q0.z; dq[3]=q0.w; dq[4]=q1.x; dq[5]=q1.y; dq[6]=q1.z; dq[7]=q1.w;
        u32* dk = (u32*)&sk[tid * 16];
        dk[0]=k0.x; dk[1]=k0.y; dk[2]=k0.z; dk[3]=k0.w; dk[4]=k1.x; dk[5]=k1.y; dk[6]=k1.z; dk[7]=k1.w;
        u32* dv = (u32*)&sv[tid * 16];
        dv[0]=v0.x; dv[1]=v0.y; dv[2]=v0.z; dv[3]=v0.w; dv[4]=v1.x; dv[5]=v1.y; dv[6]=v1.z; dv[7]=v1.w;
    }
    __syncthreads();
    {
        int s = tid >> 3, t = tid & 7;
        float sum = 0.f;
        for (int j = 0; j < 128; ++j) sum += bf2f(sq[s * 128 + j]) * bf2f(sk[t * 128 + j]);
        // scale = (dph // H)^-0.5 = 1/sqrt(32); bias chunk = b2 % BQ (tile semantics)
        sc[tid] = sum * 0.17677669529663687f
                + biasL[(size_t)(blockIdx.x & (N_BQ - 1)) * 64 + tid];
    }
    __syncthreads();
    if (tid < 8) {
        float* row = &sc[tid * 8];
        float m = row[0];
        for (int j = 1; j < 8; ++j) m = fmaxf(m, row[j]);
        float s = 0.f;
        for (int j = 0; j < 8; ++j) { float e = expf(row[j] - m); row[j] = e; s += e; }
        float inv = 1.0f / s;
        for (int j = 0; j < 8; ++j) row[j] *= inv;
    }
    __syncthreads();
    for (int idx = tid; idx < 1024; idx += 64) {
        int s = idx >> 7, e = idx & 127;
        float a = 0.f;
#pragma unroll
        for (int t = 0; t < 8; ++t) a += sc[s * 8 + t] * bf2f(sv[t * 128 + e]);
        Ctx[base + idx] = f2bf(a);
    }
}

// LayerNorm over D=512, one block (256 thr, 2 elems/thr) per row
__launch_bounds__(256)
__global__ void k_layernorm(const u16* __restrict__ in, const float* __restrict__ g,
                            const float* __restrict__ b, u16* __restrict__ out)
{
    __shared__ float red[256];
    const int tid = threadIdx.x;
    const size_t row = blockIdx.x;
    const u16* x = in + row * N_D;
    float v0 = bf2f(x[tid]), v1 = bf2f(x[tid + 256]);
    red[tid] = v0 + v1;
    __syncthreads();
    for (int off = 128; off > 0; off >>= 1) {
        if (tid < off) red[tid] += red[tid + off];
        __syncthreads();
    }
    float mu = red[0] * (1.0f / N_D);
    __syncthreads();
    float d0 = v0 - mu, d1 = v1 - mu;
    red[tid] = d0 * d0 + d1 * d1;
    __syncthreads();
    for (int off = 128; off > 0; off >>= 1) {
        if (tid < off) red[tid] += red[tid + off];
        __syncthreads();
    }
    float inv = rsqrtf(red[0] * (1.0f / N_D) + 1e-5f);
    u16* y = out + row * N_D;
    y[tid]       = f2bf(g[tid]       * d0 * inv + b[tid]);
    y[tid + 256] = f2bf(g[tid + 256] * d1 * inv + b[tid + 256]);
}

__global__ void k_zero_stats(float* stats)
{
    if (threadIdx.x < 2 && blockIdx.x == 0) stats[threadIdx.x] = 0.f;
}

__launch_bounds__(256)
__global__ void k_bn_partial(const float* __restrict__ H0, float* __restrict__ stats)
{
    __shared__ float r0[256], r1[256];
    int tid = threadIdx.x;
    float s = 0.f, s2 = 0.f;
    for (size_t i = (size_t)blockIdx.x * 256 + tid; i < (size_t)N_BQ * N_MD;
         i += (size_t)gridDim.x * 256) {
        float v = H0[i]; s += v; s2 += v * v;
    }
    r0[tid] = s; r1[tid] = s2;
    __syncthreads();
    for (int off = 128; off > 0; off >>= 1) {
        if (tid < off) { r0[tid] += r0[tid + off]; r1[tid] += r1[tid + off]; }
        __syncthreads();
    }
    if (tid == 0) { atomicAdd(&stats[0], r0[0]); atomicAdd(&stats[1], r1[0]); }
}

__global__ void k_bn_final(const float* __restrict__ H0, const float* __restrict__ stats,
                           const float* __restrict__ bn_g, const float* __restrict__ bn_b,
                           float* __restrict__ out)
{
    int i = blockIdx.x * blockDim.x + threadIdx.x;
    if (i >= N_BQ * N_MD) return;
    const float n = (float)(N_BQ * N_MD);
    float mu = stats[0] / n;
    float var = stats[1] / n - mu * mu;
    float v = bn_g[0] * (H0[i] - mu) * rsqrtf(var + 1e-5f) + bn_b[0];
    out[i] = (v >= 0.f) ? v : 0.01f * v;   // LeakyReLU; output = h[:,0,:]
}

// ---------------------------------------------------------------------
extern "C" void kernel_launch(void* const* d_in, const int* in_sizes, int n_in,
                              void* d_out, int out_size, void* d_ws, size_t ws_size,
                              hipStream_t stream)
{
    (void)in_sizes; (void)n_in; (void)out_size; (void)ws_size;
    const float* support_emb = (const float*)d_in[0];
    const float* query_t  = (const float*)d_in[1];
    const float* support_t= (const float*)d_in[2];
    const float* cls_emb  = (const float*)d_in[3];
    const float* t_freq   = (const float*)d_in[4];
    const float* t_phase  = (const float*)d_in[5];
    const float* Wq = (const float*)d_in[6];   const float* bq = (const float*)d_in[7];
    const float* Wk = (const float*)d_in[8];   const float* bk = (const float*)d_in[9];
    const float* Wv = (const float*)d_in[10];  const float* bv = (const float*)d_in[11];
    const float* Wo = (const float*)d_in[12];  const float* bo = (const float*)d_in[13];
    const float* ln1_g = (const float*)d_in[14]; const float* ln1_b = (const float*)d_in[15];
    const float* time_w = (const float*)d_in[16];
    const float* W1 = (const float*)d_in[17];  const float* b1 = (const float*)d_in[18];
    const float* W2 = (const float*)d_in[19];  const float* b2 = (const float*)d_in[20];
    const float* ln2_g = (const float*)d_in[21]; const float* ln2_b = (const float*)d_in[22];
    const float* Wc = (const float*)d_in[23];  const float* bc = (const float*)d_in[24];
    const float* bn_g = (const float*)d_in[25]; const float* bn_b = (const float*)d_in[26];

    char* ws = (char*)d_ws;
    size_t off = 0;
    auto take = [&](size_t bytes) -> char* {
        char* p = ws + off;
        off += (bytes + 255) & ~(size_t)255;
        return p;
    };
    const size_t DD = (size_t)N_D * N_D, DF = (size_t)N_D * N_FFN;
    u16* wQ = (u16*)take(N_L * DD * 2);   // all stored transposed: [N][K] bf16
    u16* wK = (u16*)take(N_L * DD * 2);
    u16* wV = (u16*)take(N_L * DD * 2);
    u16* wO = (u16*)take(N_L * DD * 2);
    u16* w1 = (u16*)take(N_L * DF * 2);   // [FFN][D]
    u16* w2 = (u16*)take(N_L * DF * 2);   // [D][FFN]
    u16* wC = (u16*)take((size_t)N_D * N_MD * 2);  // [MD][D]
    const size_t XSZ = (size_t)N_ROWS * N_D;
    u16* X    = (u16*)take(XSZ * 2);
    u16* QB   = (u16*)take(XSZ * 2);
    u16* KB   = (u16*)take(XSZ * 2);
    u16* VB   = (u16*)take(XSZ * 2);
    u16* CTX  = (u16*)take(XSZ * 2);
    u16* OUTB = (u16*)take(XSZ * 2);
    u16* T2   = (u16*)take(XSZ * 2);
    u16* T1   = (u16*)take((size_t)N_ROWS * N_FFN * 2);
    float* BIAS = (float*)take((size_t)N_L * N_BQ * 64 * 4);
    float* H0   = (float*)take((size_t)N_BQ * N_MD * 4);
    float* STAT = (float*)take(256);

    const dim3 blk(256);
    // weights -> transposed bf16 ([K][N] f32 -> [N][K] bf16), per launch
    {
        int n = (int)(N_L * DD);
        k_w_transpose_bf16<<<dim3((n + 255) / 256), blk, 0, stream>>>(Wq, wQ, N_D, N_D, n);
        k_w_transpose_bf16<<<dim3((n + 255) / 256), blk, 0, stream>>>(Wk, wK, N_D, N_D, n);
        k_w_transpose_bf16<<<dim3((n + 255) / 256), blk, 0, stream>>>(Wv, wV, N_D, N_D, n);
        k_w_transpose_bf16<<<dim3((n + 255) / 256), blk, 0, stream>>>(Wo, wO, N_D, N_D, n);
        int nf = (int)(N_L * DF);
        k_w_transpose_bf16<<<dim3((nf + 255) / 256), blk, 0, stream>>>(W1, w1, N_D, N_FFN, nf);
        k_w_transpose_bf16<<<dim3((nf + 255) / 256), blk, 0, stream>>>(W2, w2, N_FFN, N_D, nf);
        int nc = N_D * N_MD;
        k_w_transpose_bf16<<<dim3((nc + 255) / 256), blk, 0, stream>>>(Wc, wC, N_D, N_MD, nc);
    }
    // x0 and per-layer attention bias
    k_build_x0<<<dim3((unsigned)((XSZ + 255) / 256)), blk, 0, stream>>>(support_emb, cls_emb, X);
    k_bias<<<dim3(N_BQ * 64 / 256), blk, 0, stream>>>(query_t, support_t, t_freq, t_phase, time_w, BIAS);

    const dim3 gD(N_D / 64, N_ROWS / 128);      // N=512 GEMMs
    const dim3 gF(N_FFN / 64, N_ROWS / 128);    // N=2048 GEMM
    for (int l = 0; l < N_L; ++l) {
        const u16* wq_l = wQ + l * DD; const u16* wk_l = wK + l * DD;
        const u16* wv_l = wV + l * DD; const u16* wo_l = wO + l * DD;
        const u16* w1_l = w1 + l * DF; const u16* w2_l = w2 + l * DF;

        gemm_bf16_wmma<0, false, false><<<gD, blk, 0, stream>>>(
            X, N_D, wq_l, N_D, bq + l * N_D, (const u16*)nullptr, 0,
            (void*)QB, N_D, N_ROWS, N_D, N_D);
        gemm_bf16_wmma<0, false, false><<<gD, blk, 0, stream>>>(
            X, N_D, wk_l, N_D, bk + l * N_D, (const u16*)nullptr, 0,
            (void*)KB, N_D, N_ROWS, N_D, N_D);
        gemm_bf16_wmma<0, false, false><<<gD, blk, 0, stream>>>(
            X, N_D, wv_l, N_D, bv + l * N_D, (const u16*)nullptr, 0,
            (void*)VB, N_D, N_ROWS, N_D, N_D);

        k_attention<<<dim3(N_BQ * N_H), dim3(64), 0, stream>>>(
            QB, KB, VB, BIAS + (size_t)l * N_BQ * 64, CTX);

        // out-proj + residual, then LN1
        gemm_bf16_wmma<0, true, false><<<gD, blk, 0, stream>>>(
            CTX, N_D, wo_l, N_D, bo + l * N_D, X, N_D,
            (void*)T2, N_D, N_ROWS, N_D, N_D);
        k_layernorm<<<dim3(N_ROWS), blk, 0, stream>>>(T2, ln1_g + l * N_D, ln1_b + l * N_D, OUTB);

        // FFN: gelu(out@W1+b1) @ W2 + b2 + out, then LN2 -> X
        gemm_bf16_wmma<1, false, false><<<gF, blk, 0, stream>>>(
            OUTB, N_D, w1_l, N_D, b1 + l * N_FFN, (const u16*)nullptr, 0,
            (void*)T1, N_FFN, N_ROWS, N_FFN, N_D);
        gemm_bf16_wmma<0, true, false><<<gD, blk, 0, stream>>>(
            T1, N_FFN, w2_l, N_FFN, b2 + l * N_D, OUTB, N_D,
            (void*)T2, N_D, N_ROWS, N_D, N_FFN);
        k_layernorm<<<dim3(N_ROWS), blk, 0, stream>>>(T2, ln2_g + l * N_D, ln2_b + l * N_D, X);
    }

    // compress: only s==0 rows are needed for the output (BN stats are per-s)
    gemm_bf16_wmma<0, false, true><<<dim3(N_MD / 64, N_BQ / 128), blk, 0, stream>>>(
        X, N_S * N_D, wC, N_D, bc, (const u16*)nullptr, 0,
        (void*)H0, N_MD, N_BQ, N_MD, N_D);

    k_zero_stats<<<dim3(1), dim3(32), 0, stream>>>(STAT);
    k_bn_partial<<<dim3(512), blk, 0, stream>>>(H0, STAT);
    k_bn_final<<<dim3(N_BQ * N_MD / 256), blk, 0, stream>>>(H0, STAT, bn_g, bn_b, (float*)d_out);
}
